// HGRNAttention_79877801771388
// MI455X (gfx1250) — matrix operand order repaired and checked by reading, not compiled, not run on hardware
//
#include <hip/hip_runtime.h>
#include <hip/hip_bf16.h>
#include <cstdint>
#include <cstddef>

// ---------------------------------------------------------------------------
// HGRN attention block for MI455X (gfx1250, wave32, WMMA, async LDS copies).
// Pipeline:
//   K0  convert x -> bf16 ; transpose Wi/Wf/Wg/Wo -> bf16 [out,in]
//   K1  fused 3-way WMMA GEMM (i,f,g), block tile 128x64, wave tile 32x32,
//       double-buffered async LDS staging (ASYNCcnt path)
//   K2  chunked parallel scan h_t = a_t h_{t-1} + i_t (3 passes, coalesced)
//   K3  RMSNorm * w * silu(g) -> bf16
//   K4  WMMA GEMM (o_norm @ Wo) -> f32 output, same structure
// ---------------------------------------------------------------------------

#define DDIM 2048
#define EPS_RMS 1e-5f

typedef __bf16 bf16_t;
typedef __attribute__((ext_vector_type(16))) __bf16 v16bf;
typedef __attribute__((ext_vector_type(8)))  float  v8f;

__device__ __forceinline__ bf16_t f2bf(float f) {
    union { float f; uint32_t u; } in; in.f = f;
    uint32_t u = in.u;
    u += 0x7fffu + ((u >> 16) & 1u);       // round-to-nearest-even
    union { uint16_t s; bf16_t b; } out;
    out.s = (uint16_t)(u >> 16);
    return out.b;
}

__device__ __forceinline__ float sigmoid_f(float x) {
    return 1.0f / (1.0f + __expf(-x));
}

// 16B global -> LDS async copy (CDNA5 ASYNCcnt path, GV addressing).
#define ASYNC_CP16(ldsPtr, gPtr)                                              \
    asm volatile("global_load_async_to_lds_b128 %0, %1, off"                  \
                 :: "v"((uint32_t)(uintptr_t)(ldsPtr)),                       \
                    "v"((const void*)(gPtr))                                  \
                 : "memory")

#define WAIT_ASYNC0() asm volatile("s_wait_asynccnt 0x0" ::: "memory")

// Build a 16x32 bf16 B fragment from an LDS tile stored [n][k] (32 halves/row).
#define LDS_BFRAG(sBm_, nloc_, qo_)                                           \
    ({ union { v16bf v; uint4 q[2]; } _ub;                                    \
       _ub.q[0] = (sBm_)[(nloc_) * 4 + (qo_)];                                \
       _ub.q[1] = (sBm_)[(nloc_) * 4 + (qo_) + 2];                            \
       _ub.v; })

#define WMMA_BF16(a_, b_, c_)                                                 \
    __builtin_amdgcn_wmma_f32_16x16x32_bf16(false, (a_), false, (b_),         \
                                            (short)0, (c_), false, false)

// ---------------------------------------------------------------- K0 helpers
__global__ void cvt_f32_bf16_kernel(const float* __restrict__ in,
                                    bf16_t* __restrict__ out, int n) {
    int i = blockIdx.x * blockDim.x + threadIdx.x;
    int stride = gridDim.x * blockDim.x;
    for (; i < n; i += stride) out[i] = f2bf(in[i]);
}

// W is [in=D, out=D] row-major f32; produce WT bf16 [out=D, in=D] row-major.
__global__ void transpose_w_bf16_kernel(const float* __restrict__ W,
                                        bf16_t* __restrict__ WT) {
    int i = blockIdx.x * blockDim.x + threadIdx.x;
    int stride = gridDim.x * blockDim.x;
    const int n = DDIM * DDIM;
    for (; i < n; i += stride) {
        int e = i / DDIM, d = i - e * DDIM;
        WT[(size_t)e * DDIM + d] = f2bf(W[(size_t)d * DDIM + e]);
    }
}

// -------------------------------------------------------------- K1: proj3
// Block: 256 threads = 8 waves; block tile 128(M) x 64(N); wave tile 32 x 32.
// A and all three B tiles double-buffered in LDS via async copies.
__global__ __launch_bounds__(256) void proj3_wmma_kernel(
    const bf16_t* __restrict__ xb,    // [rows, D] bf16
    const bf16_t* __restrict__ WiT,   // [D, D] bf16 (out-major)
    const bf16_t* __restrict__ WfT,
    const bf16_t* __restrict__ WgT,
    float* __restrict__ aBuf,         // forget gate  sigmoid(f)
    float* __restrict__ ioBuf,        // gated input  silu(i)*(1-sigmoid(f))
    float* __restrict__ gBuf)         // raw g
{
    __shared__ uint4 sA[2][512];      // [buf][128 rows x 32 K bf16]  8 KB each
    __shared__ uint4 sB[2][3][256];   // [buf][mat][64 cols x 32 K]   4 KB each

    const int tid  = threadIdx.x;
    const int lane = tid & 31;
    const int wave = tid >> 5;
    const int wm   = wave & 3;        // 4 M sub-tiles of 32
    const int wn   = wave >> 2;       // 2 N sub-tiles of 32

    const int rowBase  = blockIdx.x * 128;
    const int colBase0 = blockIdx.y * 64;
    const int colBase  = colBase0 + wn * 32;

    const int hb   = (lane < 16) ? 0 : 1;    // A K-chunk select (uint4 units)
    const int qo   = (lane < 16) ? 0 : 1;    // B K-base select  (uint4 units)
    const int nl0  = wn * 32 + (lane & 15);  // B column (local), tile 0

    // Staging: A 128x32 halves -> 2 x b128 per thread; B 64x32 -> 1 each.
    const bf16_t* gA = xb  + (size_t)(rowBase  + (tid >> 1)) * DDIM
                           + (tid & 1) * 16;
    const bf16_t* gI = WiT + (size_t)(colBase0 + (tid >> 2)) * DDIM
                           + (tid & 3) * 8;
    const bf16_t* gF = WfT + (size_t)(colBase0 + (tid >> 2)) * DDIM
                           + (tid & 3) * 8;
    const bf16_t* gG = WgT + (size_t)(colBase0 + (tid >> 2)) * DDIM
                           + (tid & 3) * 8;

    auto stage = [&](int b, int k0s) {
        ASYNC_CP16(&sA[b][2 * tid],     gA + k0s);
        ASYNC_CP16(&sA[b][2 * tid + 1], gA + k0s + 8);
        ASYNC_CP16(&sB[b][0][tid],      gI + k0s);
        ASYNC_CP16(&sB[b][1][tid],      gF + k0s);
        ASYNC_CP16(&sB[b][2][tid],      gG + k0s);
    };

    v8f acc[3][2][2] = {};            // [mat: i,f,g][mtile][ntile]

    stage(0, 0);
    int buf = 0;
    for (int k0 = 0; k0 < DDIM; k0 += 32) {
        WAIT_ASYNC0();                // my async copies have landed in LDS
        __syncthreads();              // everyone's copies landed; prev reads done
        if (k0 + 32 < DDIM) stage(buf ^ 1, k0 + 32);

        union { v16bf v; uint4 q[2]; } uaA[2];   // two 16x32 A fragments
#pragma unroll
        for (int mt = 0; mt < 2; ++mt) {
            const int mloc = wm * 32 + mt * 16 + (lane & 15);
            uaA[mt].q[0] = sA[buf][mloc * 4 + hb];
            uaA[mt].q[1] = sA[buf][mloc * 4 + hb + 2];
        }
#pragma unroll
        for (int m = 0; m < 3; ++m) {
            const uint4* sBp = sB[buf][m];
            v16bf b0 = LDS_BFRAG(sBp, nl0,      qo);
            v16bf b1 = LDS_BFRAG(sBp, nl0 + 16, qo);
            acc[m][0][0] = WMMA_BF16(uaA[0].v, b0, acc[m][0][0]);
            acc[m][1][0] = WMMA_BF16(uaA[1].v, b0, acc[m][1][0]);
            acc[m][0][1] = WMMA_BF16(uaA[0].v, b1, acc[m][0][1]);
            acc[m][1][1] = WMMA_BF16(uaA[1].v, b1, acc[m][1][1]);
        }
        buf ^= 1;
    }

    // Epilogue: C layout -> row = r + (lane<16?0:8), col = lane%16.
#pragma unroll
    for (int mt = 0; mt < 2; ++mt) {
        const int mg = rowBase + wm * 32 + mt * 16 + ((lane < 16) ? 0 : 8);
#pragma unroll
        for (int nt = 0; nt < 2; ++nt) {
            const int cg = colBase + nt * 16 + (lane & 15);
#pragma unroll
            for (int r = 0; r < 8; ++r) {
                size_t o = (size_t)(mg + r) * DDIM + cg;
                float ir = acc[0][mt][nt][r];
                float fr = acc[1][mt][nt][r];
                float a  = sigmoid_f(fr);
                float iv = ir * sigmoid_f(ir) * (1.0f - a);
                aBuf[o] = a; ioBuf[o] = iv; gBuf[o] = acc[2][mt][nt][r];
            }
        }
    }
}

// -------------------------------------------------------------- K2: scan
// 3-pass chunked scan for parallelism: (Bn*NC*D) threads for passes 1 and 3.
// Pass 1: local scan per chunk (h0=0), record chunk product & final h.
__global__ void scan_pass1_kernel(const float* __restrict__ aBuf,
                                  float* __restrict__ ioBuf, // in: i, out: lh
                                  float* __restrict__ aP,    // [Bn*NC*D]
                                  float* __restrict__ hE,    // [Bn*NC*D]
                                  int T, int CH, int NC) {
    int id = blockIdx.x * blockDim.x + threadIdx.x;   // (b, c, d)
    int d  = id % DDIM;
    int c  = (id / DDIM) % NC;
    int b  = id / (DDIM * NC);
    size_t base = ((size_t)b * T + (size_t)c * CH) * DDIM + d;
    float h = 0.0f, p = 1.0f;
    for (int t = 0; t < CH; ++t) {
        size_t idx = base + (size_t)t * DDIM;
        float av = aBuf[idx];
        h = fmaf(av, h, ioBuf[idx]);
        p *= av;
        ioBuf[idx] = h;
    }
    aP[id] = p;
    hE[id] = h;
}

// Pass 2: sequential combine of NC chunk carries per channel (tiny).
__global__ void scan_pass2_kernel(const float* __restrict__ aP,
                                  const float* __restrict__ hE,
                                  float* __restrict__ cin,   // [Bn*NC*D]
                                  int NC) {
    int ch = blockIdx.x * blockDim.x + threadIdx.x;   // (b, d)
    int d  = ch % DDIM;
    int b  = ch / DDIM;
    float carry = 0.0f;
    for (int c = 0; c < NC; ++c) {
        size_t idx = ((size_t)b * NC + c) * DDIM + d;
        cin[idx] = carry;
        carry = fmaf(aP[idx], carry, hE[idx]);
    }
}

// Pass 3: h_t = lh_t + (prod_{s<=t} a_s) * carry_in.
__global__ void scan_pass3_kernel(const float* __restrict__ aBuf,
                                  float* __restrict__ ioBuf, // in: lh, out: o
                                  const float* __restrict__ cin,
                                  int T, int CH, int NC) {
    int id = blockIdx.x * blockDim.x + threadIdx.x;   // (b, c, d)
    int d  = id % DDIM;
    int c  = (id / DDIM) % NC;
    int b  = id / (DDIM * NC);
    float carry = cin[id];
    size_t base = ((size_t)b * T + (size_t)c * CH) * DDIM + d;
    float p = 1.0f;
    for (int t = 0; t < CH; ++t) {
        size_t idx = base + (size_t)t * DDIM;
        p *= aBuf[idx];
        ioBuf[idx] = fmaf(p, carry, ioBuf[idx]);
    }
}

// -------------------------------------------------------------- K3: norm
__global__ __launch_bounds__(256) void rmsnorm_gate_kernel(
    const float* __restrict__ oBuf, const float* __restrict__ gBuf,
    const float* __restrict__ w, bf16_t* __restrict__ outb) {
    int row = blockIdx.x;
    const float* op = oBuf + (size_t)row * DDIM;
    const float* gp = gBuf + (size_t)row * DDIM;

    float ss = 0.0f;
    for (int d = threadIdx.x; d < DDIM; d += 256) { float v = op[d]; ss += v * v; }
    __shared__ float red[256];
    red[threadIdx.x] = ss;
    __syncthreads();
    for (int s = 128; s > 0; s >>= 1) {
        if (threadIdx.x < s) red[threadIdx.x] += red[threadIdx.x + s];
        __syncthreads();
    }
    float rinv = rsqrtf(red[0] * (1.0f / DDIM) + EPS_RMS);

    for (int d = threadIdx.x; d < DDIM; d += 256) {
        float gv = gp[d];
        float v  = op[d] * rinv * w[d] * gv * sigmoid_f(gv);
        outb[(size_t)row * DDIM + d] = f2bf(v);
    }
}

// -------------------------------------------------------------- K4: out GEMM
__global__ __launch_bounds__(256) void gemm_out_wmma_kernel(
    const bf16_t* __restrict__ ob,    // [rows, D] bf16 (normed, gated)
    const bf16_t* __restrict__ WoT,   // [D, D] bf16 (out-major)
    float* __restrict__ out)          // [rows, D] f32
{
    __shared__ uint4 sA[2][512];
    __shared__ uint4 sBo[2][256];

    const int tid  = threadIdx.x;
    const int lane = tid & 31;
    const int wave = tid >> 5;
    const int wm   = wave & 3;
    const int wn   = wave >> 2;

    const int rowBase  = blockIdx.x * 128;
    const int colBase0 = blockIdx.y * 64;
    const int colBase  = colBase0 + wn * 32;

    const int hb   = (lane < 16) ? 0 : 1;
    const int qo   = (lane < 16) ? 0 : 1;
    const int nl0  = wn * 32 + (lane & 15);

    const bf16_t* gA = ob  + (size_t)(rowBase  + (tid >> 1)) * DDIM
                           + (tid & 1) * 16;
    const bf16_t* gW = WoT + (size_t)(colBase0 + (tid >> 2)) * DDIM
                           + (tid & 3) * 8;

    auto stage = [&](int b, int k0s) {
        ASYNC_CP16(&sA[b][2 * tid],     gA + k0s);
        ASYNC_CP16(&sA[b][2 * tid + 1], gA + k0s + 8);
        ASYNC_CP16(&sBo[b][tid],        gW + k0s);
    };

    v8f acc[2][2] = {};               // [mtile][ntile]

    stage(0, 0);
    int buf = 0;
    for (int k0 = 0; k0 < DDIM; k0 += 32) {
        WAIT_ASYNC0();
        __syncthreads();
        if (k0 + 32 < DDIM) stage(buf ^ 1, k0 + 32);

        union { v16bf v; uint4 q[2]; } uaA[2];
#pragma unroll
        for (int mt = 0; mt < 2; ++mt) {
            const int mloc = wm * 32 + mt * 16 + (lane & 15);
            uaA[mt].q[0] = sA[buf][mloc * 4 + hb];
            uaA[mt].q[1] = sA[buf][mloc * 4 + hb + 2];
        }
        v16bf b0 = LDS_BFRAG(sBo[buf], nl0,      qo);
        v16bf b1 = LDS_BFRAG(sBo[buf], nl0 + 16, qo);
        acc[0][0] = WMMA_BF16(uaA[0].v, b0, acc[0][0]);
        acc[1][0] = WMMA_BF16(uaA[1].v, b0, acc[1][0]);
        acc[0][1] = WMMA_BF16(uaA[0].v, b1, acc[0][1]);
        acc[1][1] = WMMA_BF16(uaA[1].v, b1, acc[1][1]);
        buf ^= 1;
    }

#pragma unroll
    for (int mt = 0; mt < 2; ++mt) {
        const int mg = rowBase + wm * 32 + mt * 16 + ((lane < 16) ? 0 : 8);
#pragma unroll
        for (int nt = 0; nt < 2; ++nt) {
            const int cg = colBase + nt * 16 + (lane & 15);
#pragma unroll
            for (int r = 0; r < 8; ++r)
                out[(size_t)(mg + r) * DDIM + cg] = acc[mt][nt][r];
        }
    }
}

// ---------------------------------------------------------------------------
extern "C" void kernel_launch(void* const* d_in, const int* in_sizes, int n_in,
                              void* d_out, int out_size, void* d_ws, size_t ws_size,
                              hipStream_t stream) {
    const float* x   = (const float*)d_in[0];   // [B,T,D]
    const float* Wi  = (const float*)d_in[1];
    const float* Wf  = (const float*)d_in[2];
    const float* Wg  = (const float*)d_in[3];
    const float* Wo  = (const float*)d_in[4];
    const float* gw  = (const float*)d_in[5];
    float* out = (float*)d_out;

    const int rows = in_sizes[0] / DDIM;        // B*T = 16384
    const int T    = (rows % 4096 == 0) ? 4096 : rows;
    const int Bn   = rows / T;
    const int NC   = (T % 16 == 0) ? 16 : 1;    // scan time-chunks
    const int CH   = T / NC;
    const size_t nElem = (size_t)rows * DDIM;
    const size_t wElem = (size_t)DDIM * DDIM;

    // ---- carve workspace (~560 MB total) ----
    char* p = (char*)d_ws;
    auto carve = [&](size_t bytes) {
        void* r = (void*)p;
        p += (bytes + 255) & ~(size_t)255;
        return r;
    };
    bf16_t* xb   = (bf16_t*)carve(nElem * 2);
    bf16_t* WiT  = (bf16_t*)carve(wElem * 2);
    bf16_t* WfT  = (bf16_t*)carve(wElem * 2);
    bf16_t* WgT  = (bf16_t*)carve(wElem * 2);
    bf16_t* WoT  = (bf16_t*)carve(wElem * 2);
    float*  aBuf = (float*)carve(nElem * 4);
    float*  ioBuf= (float*)carve(nElem * 4);
    float*  gBuf = (float*)carve(nElem * 4);
    bf16_t* onb  = (bf16_t*)carve(nElem * 2);
    float*  aP   = (float*)carve((size_t)Bn * NC * DDIM * 4);
    float*  hE   = (float*)carve((size_t)Bn * NC * DDIM * 4);
    float*  cin  = (float*)carve((size_t)Bn * NC * DDIM * 4);

    // K0: precision conversion / weight transpose (weights L2-resident after)
    cvt_f32_bf16_kernel<<<4096, 256, 0, stream>>>(x, xb, (int)nElem);
    transpose_w_bf16_kernel<<<2048, 256, 0, stream>>>(Wi, WiT);
    transpose_w_bf16_kernel<<<2048, 256, 0, stream>>>(Wf, WfT);
    transpose_w_bf16_kernel<<<2048, 256, 0, stream>>>(Wg, WgT);
    transpose_w_bf16_kernel<<<2048, 256, 0, stream>>>(Wo, WoT);

    // K1: fused i/f/g projections + gates (bf16 WMMA, async LDS pipeline)
    dim3 gGemm(rows / 128, DDIM / 64);
    proj3_wmma_kernel<<<gGemm, 256, 0, stream>>>(xb, WiT, WfT, WgT,
                                                 aBuf, ioBuf, gBuf);

    // K2: chunked parallel linear recurrence over time
    int nScan = Bn * NC * DDIM;
    scan_pass1_kernel<<<nScan / 256, 256, 0, stream>>>(aBuf, ioBuf, aP, hE,
                                                       T, CH, NC);
    int channels = Bn * DDIM;
    scan_pass2_kernel<<<channels / 256, 256, 0, stream>>>(aP, hE, cin, NC);
    scan_pass3_kernel<<<nScan / 256, 256, 0, stream>>>(aBuf, ioBuf, cin,
                                                       T, CH, NC);

    // K3: RMSNorm * weight * silu(g) -> bf16
    rmsnorm_gate_kernel<<<rows, 256, 0, stream>>>(ioBuf, gBuf, gw, onb);

    // K4: output projection
    gemm_out_wmma_kernel<<<gGemm, 256, 0, stream>>>(onb, WoT, out);
}
